// TemplateDistribution_WordsSoftDistribution_36215164240853
// MI455X (gfx1250) — matrix operand order repaired and checked by reading, not compiled, not run on hardware
//
#include <hip/hip_runtime.h>

// ---------------- constants ----------------
#define DIMc 1024
#define Bc   2048
#define Sc   77
#define Tc   16
#define NTc  64

typedef __attribute__((ext_vector_type(16))) __bf16 v16bf;
typedef __attribute__((ext_vector_type(8)))  __bf16 v8bf;
typedef __attribute__((ext_vector_type(4)))  __bf16 v4bf;
typedef __attribute__((ext_vector_type(8)))  float  v8f;

static __device__ __forceinline__ v8f zero8() {
    v8f z;
#pragma unroll
    for (int i = 0; i < 8; ++i) z[i] = 0.f;
    return z;
}

union frag16 { v16bf v; v8bf h[2]; };

// A fragment (16x32 bf16, MxK): this lane holds row m = lane&15.
// element i -> k = k0 + (i&7) + 8*half + 16*(i>=8)
static __device__ __forceinline__ v16bf a_frag(const __bf16* rowp, int k0, int half) {
    frag16 u;
    u.h[0] = *(const v8bf*)(rowp + k0 + half * 8);
    u.h[1] = *(const v8bf*)(rowp + k0 + 16 + half * 8);
    return u.v;
}

// B fragment (32x16 bf16, KxN) with B[k][n] = W[n][k]:
// this lane holds column n = lane&15; element i -> k = k0 + (i&7) + 8*half + 16*(i>=8)
static __device__ __forceinline__ v16bf b_frag(const __bf16* wrowp, int k0, int half) {
    frag16 u;
    u.h[0] = *(const v8bf*)(wrowp + k0 + half * 8);
    u.h[1] = *(const v8bf*)(wrowp + k0 + 16 + half * 8);
    return u.v;
}

static __device__ __forceinline__ v8f wmma_bf16(v16bf a, v16bf b, v8f c) {
    return __builtin_amdgcn_wmma_f32_16x16x32_bf16(false, a, false, b, (short)0, c, false, false);
}

// ---------------- fp32 -> bf16 conversion (hardware cvt) ----------------
__global__ __launch_bounds__(256) void k_cvt(const float* __restrict__ src,
                                             __bf16* __restrict__ dst, int n) {
    int i = (blockIdx.x * 256 + threadIdx.x) * 4;
    if (i < n) {
        float4 f = *(const float4*)(src + i);
        v4bf o;
        o[0] = (__bf16)f.x; o[1] = (__bf16)f.y; o[2] = (__bf16)f.z; o[3] = (__bf16)f.w;
        *(v4bf*)(dst + i) = o;
    }
}

// ---------------- grouping kernels ----------------
__global__ void k_init(int* counts, int* cursor) {
    int t = threadIdx.x;
    if (t < Tc) { counts[t] = 0; cursor[t] = 0; }
}

__global__ void k_hist(const int* ts, int* counts) {
    int b = blockIdx.x * blockDim.x + threadIdx.x;
    if (b < Bc) atomicAdd(&counts[ts[b]], 1);
}

__global__ void k_scan(const int* counts, int* offs) {
    if (threadIdx.x == 0) {
        int a = 0;
        for (int t = 0; t < Tc; ++t) { offs[t] = a; a += counts[t]; }
    }
}

__global__ void k_scatter(const int* ts, const int* offs, int* cursor, int* perm) {
    int b = blockIdx.x * blockDim.x + threadIdx.x;
    if (b < Bc) {
        int t = ts[b];
        int p = atomicAdd(&cursor[t], 1);
        perm[offs[t] + p] = b;
    }
}

// ---------------- x1: grouped gather-GEMM ----------------
// x1[b,:] = W1_w[t_b] @ tsf[b,:] + W1_b[t_b]   (output in bf16)
// grid (Tc, 2); block 256 = 8 waves; wave covers 64 cols.
__global__ __launch_bounds__(256) void k_x1(const __bf16* __restrict__ tsf,
                                            const __bf16* __restrict__ W1w,
                                            const float* __restrict__ W1b,
                                            const int* __restrict__ counts,
                                            const int* __restrict__ offs,
                                            const int* __restrict__ perm,
                                            __bf16* __restrict__ x1) {
    int g = blockIdx.x, slab = blockIdx.y;
    int wave = threadIdx.x >> 5, lane = threadIdx.x & 31;
    int n_lane = lane & 15, half = lane >> 4;
    int cnt = counts[g], off = offs[g];
    int colBase = slab * 512 + wave * 64;
    const __bf16* Wg = W1w + (size_t)g * DIMc * DIMc;
    const __bf16* wrow[4];
#pragma unroll
    for (int j = 0; j < 4; ++j)
        wrow[j] = Wg + (size_t)(colBase + j * 16 + n_lane) * DIMc;

    int nrt = (cnt + 15) >> 4;
    for (int rt = 0; rt < nrt; ++rt) {
        int ridx = rt * 16 + n_lane;
        int sload = perm[off + (ridx < cnt ? ridx : 0)];
        const __bf16* arow = tsf + (size_t)sload * DIMc;
        v8f acc[4];
#pragma unroll
        for (int j = 0; j < 4; ++j) acc[j] = zero8();
        for (int k0 = 0; k0 < DIMc; k0 += 32) {
            v16bf va = a_frag(arow, k0, half);
#pragma unroll
            for (int j = 0; j < 4; ++j) {
                v16bf vb = b_frag(wrow[j], k0, half);
                acc[j] = wmma_bf16(va, vb, acc[j]);
            }
        }
#pragma unroll
        for (int j = 0; j < 4; ++j) {
            int n = colBase + j * 16 + n_lane;
            float bb = W1b[g * DIMc + n];
#pragma unroll
            for (int r = 0; r < 8; ++r) {
                int ridx2 = rt * 16 + r + half * 8;
                if (ridx2 < cnt) {
                    int srow = perm[off + ridx2];
                    x1[(size_t)srow * DIMc + n] = (__bf16)(acc[j][r] + bb);
                }
            }
        }
    }
}

// ---------------- generic GEMM: D = act([A1|A2] @ W^T + bias) ----------------
// A/W in bf16; writes fp32 (optional) and bf16 (optional).
// grid (M/16, Ntot/512); block 256 = 8 waves; wave covers 64 cols.
__global__ __launch_bounds__(256) void k_gemm(const __bf16* __restrict__ A1,
                                              const __bf16* __restrict__ A2,
                                              int K1, int Ktot,
                                              const __bf16* __restrict__ W, int ldw,
                                              const float* __restrict__ bias,
                                              float* __restrict__ D,
                                              __bf16* __restrict__ Dbf,
                                              int ldd, int relu) {
    int rowTile = blockIdx.x;
    int wave = threadIdx.x >> 5, lane = threadIdx.x & 31;
    int n_lane = lane & 15, half = lane >> 4;
    int colBase = blockIdx.y * 512 + wave * 64;
    int m = rowTile * 16 + n_lane;
    const __bf16* a1 = A1 + (size_t)m * K1;
    const __bf16* a2 = A2 ? A2 + (size_t)m * (Ktot - K1) : nullptr;
    const __bf16* wrow[4];
#pragma unroll
    for (int j = 0; j < 4; ++j)
        wrow[j] = W + (size_t)(colBase + j * 16 + n_lane) * ldw;

    v8f acc[4];
#pragma unroll
    for (int j = 0; j < 4; ++j) acc[j] = zero8();

    for (int k0 = 0; k0 < Ktot; k0 += 32) {
        const __bf16* rp; int kk;
        if (k0 < K1) { rp = a1; kk = k0; } else { rp = a2; kk = k0 - K1; }
        v16bf va = a_frag(rp, kk, half);
#pragma unroll
        for (int j = 0; j < 4; ++j) {
            v16bf vb = b_frag(wrow[j], k0, half);
            acc[j] = wmma_bf16(va, vb, acc[j]);
        }
    }
#pragma unroll
    for (int j = 0; j < 4; ++j) {
        int n = colBase + j * 16 + n_lane;
        float bb = bias[n];
#pragma unroll
        for (int r = 0; r < 8; ++r) {
            float v = acc[j][r] + bb;
            if (relu) v = fmaxf(v, 0.f);
            size_t idx = (size_t)(rowTile * 16 + r + half * 8) * ldd + n;
            if (D)   D[idx]   = v;
            if (Dbf) Dbf[idx] = (__bf16)v;
        }
    }
}

// ---------------- head: logits (N=64) + softmax ----------------
// grid (B/16); block 128 = 4 waves (each wave one 16-col tile).
__global__ __launch_bounds__(128) void k_head(const __bf16* __restrict__ H,
                                              const __bf16* __restrict__ m3w,
                                              const float* __restrict__ m3b,
                                              float* __restrict__ out0) {
    __shared__ float tile[16][NTc + 1];
    int rowTile = blockIdx.x;
    int wave = threadIdx.x >> 5, lane = threadIdx.x & 31;
    int n_lane = lane & 15, half = lane >> 4;
    int n0 = wave * 16;
    int m = rowTile * 16 + n_lane;
    const __bf16* arow = H + (size_t)m * DIMc;
    const __bf16* wrow = m3w + (size_t)(n0 + n_lane) * DIMc;

    v8f acc = zero8();
    for (int k0 = 0; k0 < DIMc; k0 += 32) {
        v16bf va = a_frag(arow, k0, half);
        v16bf vb = b_frag(wrow, k0, half);
        acc = wmma_bf16(va, vb, acc);
    }
    float bb = m3b[n0 + n_lane];
#pragma unroll
    for (int r = 0; r < 8; ++r)
        tile[r + half * 8][n0 + n_lane] = acc[r] + bb;
    __syncthreads();

    if (threadIdx.x < 16) {
        int row = threadIdx.x;
        float mx = -3.0e38f;
        for (int j = 0; j < NTc; ++j) mx = fmaxf(mx, tile[row][j]);
        float s = 0.f;
        for (int j = 0; j < NTc; ++j) s += __expf(tile[row][j] - mx);
        float inv = 1.f / s;
        for (int j = 0; j < NTc; ++j)
            out0[(size_t)(rowTile * 16 + row) * NTc + j] = __expf(tile[row][j] - mx) * inv;
    }
}

// ---------------- fused attention: single pass over text_feats ----------------
// grid (B); block 256. Online softmax over S=77 words.
__global__ __launch_bounds__(256) void k_attn(const float* __restrict__ tf,
                                              const float* __restrict__ x,
                                              const float* __restrict__ attn_w,
                                              const float* __restrict__ attn_b,
                                              float* __restrict__ out1,
                                              float* __restrict__ out2) {
    int b = blockIdx.x;
    int tid = threadIdx.x, wid = tid >> 5, lane = tid & 31;
    __shared__ float part[8];
    __shared__ float sc[Sc];

    int c = tid * 4;
    float4 xr = *(const float4*)(x + (size_t)b * DIMc + c);
    float4 aw = *(const float4*)(attn_w + c);
    float xw0 = xr.x * aw.x, xw1 = xr.y * aw.y, xw2 = xr.z * aw.z, xw3 = xr.w * aw.w;
    float ab = attn_b[0];

    float mrun = -3.0e38f, Z = 0.f;
    float a0 = 0.f, a1 = 0.f, a2 = 0.f, a3 = 0.f;
    const float* tfb = tf + (size_t)b * Sc * DIMc;

    for (int s = 0; s < Sc; ++s) {
        const float* rowp = tfb + (size_t)s * DIMc + c;
        if (s + 1 < Sc) __builtin_prefetch(rowp + DIMc, 0, 1);
        float4 t4 = *(const float4*)rowp;
        float p = xw0 * t4.x + xw1 * t4.y + xw2 * t4.z + xw3 * t4.w;
#pragma unroll
        for (int o = 16; o > 0; o >>= 1) p += __shfl_xor(p, o, 32);
        if (lane == 0) part[wid] = p;
        __syncthreads();
        float score = part[0] + part[1] + part[2] + part[3] +
                      part[4] + part[5] + part[6] + part[7] + ab;
        __syncthreads();
        if (tid == 0) sc[s] = score;
        if (score > mrun) {
            float r = __expf(mrun - score);
            Z *= r; a0 *= r; a1 *= r; a2 *= r; a3 *= r;
            mrun = score;
        }
        float e = __expf(score - mrun);
        Z += e;
        a0 += e * t4.x; a1 += e * t4.y; a2 += e * t4.z; a3 += e * t4.w;
    }
    __syncthreads();
    float invZ = 1.f / Z;
    for (int s = tid; s < Sc; s += 256)
        out1[(size_t)b * Sc + s] = __expf(sc[s] - mrun) * invZ;
    float4 o;
    o.x = a0 * invZ; o.y = a1 * invZ; o.z = a2 * invZ; o.w = a3 * invZ;
    *(float4*)(out2 + (size_t)b * DIMc + c) = o;
}

// ---------------- launcher ----------------
extern "C" void kernel_launch(void* const* d_in, const int* in_sizes, int n_in,
                              void* d_out, int out_size, void* d_ws, size_t ws_size,
                              hipStream_t stream) {
    (void)in_sizes; (void)n_in; (void)out_size; (void)ws_size;

    const float* text_feats = (const float*)d_in[0];
    const float* tsf        = (const float*)d_in[1];
    const float* tq         = (const float*)d_in[2];
    const int*   tstep      = (const int*)d_in[3];
    const float* W1w        = (const float*)d_in[4];
    const float* W1b        = (const float*)d_in[5];
    const float* W2w        = (const float*)d_in[6];
    const float* W2b        = (const float*)d_in[7];
    const float* m1w        = (const float*)d_in[8];
    const float* m1b        = (const float*)d_in[9];
    const float* m2w        = (const float*)d_in[10];
    const float* m2b        = (const float*)d_in[11];
    const float* m3w        = (const float*)d_in[12];
    const float* m3b        = (const float*)d_in[13];
    const float* aw         = (const float*)d_in[14];
    const float* abb        = (const float*)d_in[15];

    char* ws = (char*)d_ws;
    int* counts = (int*)ws;
    int* offs   = counts + Tc;
    int* cursor = offs + Tc;
    int* perm   = cursor + Tc;

    // bf16 mirrors + activations
    __bf16* p = (__bf16*)(ws + 16384);
    __bf16* tsf_bf = p;  p += (size_t)Bc * DIMc;            // 2M
    __bf16* tq_bf  = p;  p += (size_t)Bc * DIMc;            // 2M
    __bf16* W1_bf  = p;  p += (size_t)Tc * DIMc * DIMc;     // 16M
    __bf16* W2_bf  = p;  p += (size_t)DIMc * 2 * DIMc;      // 2M
    __bf16* m1_bf  = p;  p += (size_t)DIMc * DIMc;          // 1M
    __bf16* m2_bf  = p;  p += (size_t)DIMc * DIMc;          // 1M
    __bf16* m3_bf  = p;  p += (size_t)NTc * DIMc;           // 64K
    __bf16* x1_bf  = p;  p += (size_t)Bc * DIMc;            // 2M
    __bf16* x_bf   = p;  p += (size_t)Bc * DIMc;            // 2M
    __bf16* h1_bf  = p;  p += (size_t)Bc * DIMc;            // 2M
    __bf16* h2_bf  = p;  p += (size_t)Bc * DIMc;            // 2M
    float*  x_f32  = (float*)p;                             // 8MB

    float* out0 = (float*)d_out;                        // template_distribution [B,NT]
    float* out1 = out0 + (size_t)Bc * NTc;              // words_soft_distribution [B,S]
    float* out2 = out1 + (size_t)Bc * Sc;               // c_t [B,DIM]

    // grouping
    k_init<<<1, 32, 0, stream>>>(counts, cursor);
    k_hist<<<Bc / 256, 256, 0, stream>>>(tstep, counts);
    k_scan<<<1, 32, 0, stream>>>(counts, offs);
    k_scatter<<<Bc / 256, 256, 0, stream>>>(tstep, offs, cursor, perm);

    // fp32 -> bf16 prep
    auto cvt = [&](const float* s, __bf16* d, size_t n) {
        k_cvt<<<(unsigned)((n / 4 + 255) / 256), 256, 0, stream>>>(s, d, (int)n);
    };
    cvt(tsf, tsf_bf, (size_t)Bc * DIMc);
    cvt(tq,  tq_bf,  (size_t)Bc * DIMc);
    cvt(W1w, W1_bf,  (size_t)Tc * DIMc * DIMc);
    cvt(W2w, W2_bf,  (size_t)DIMc * 2 * DIMc);
    cvt(m1w, m1_bf,  (size_t)DIMc * DIMc);
    cvt(m2w, m2_bf,  (size_t)DIMc * DIMc);
    cvt(m3w, m3_bf,  (size_t)NTc * DIMc);

    // x1 = grouped time-indexed linear (bf16 out)
    k_x1<<<dim3(Tc, 2), 256, 0, stream>>>(tsf_bf, W1_bf, W1b, counts, offs, perm, x1_bf);
    // x = concat(x1, q) @ W2^T + b2   (fp32 + bf16 out)
    k_gemm<<<dim3(Bc / 16, 2), 256, 0, stream>>>(x1_bf, tq_bf, DIMc, 2 * DIMc,
                                                 W2_bf, 2 * DIMc, W2b, x_f32, x_bf, DIMc, 0);
    // h1 = relu(x @ m1^T + b)
    k_gemm<<<dim3(Bc / 16, 2), 256, 0, stream>>>(x_bf, nullptr, DIMc, DIMc,
                                                 m1_bf, DIMc, m1b, nullptr, h1_bf, DIMc, 1);
    // h2 = relu(h1 @ m2^T + b)
    k_gemm<<<dim3(Bc / 16, 2), 256, 0, stream>>>(h1_bf, nullptr, DIMc, DIMc,
                                                 m2_bf, DIMc, m2b, nullptr, h2_bf, DIMc, 1);
    // template distribution
    k_head<<<Bc / 16, 128, 0, stream>>>(h2_bf, m3_bf, m3b, out0);
    // fused attention (uses x_f32)
    k_attn<<<Bc, 256, 0, stream>>>(text_feats, x_f32, aw, abb, out1, out2);
}
